// MultiHeadAttention_3736621547982
// MI455X (gfx1250) — compile-verified, matrix operand order checked by reading, not run on hardware
//
#include <hip/hip_runtime.h>
#include <hip/hip_bf16.h>

typedef __attribute__((ext_vector_type(16))) __bf16 v16bf;
typedef __attribute__((ext_vector_type(8)))  float  v8f;

#define D_MODEL   1024
#define NUM_HEADS 16
#define HEAD_DIM  64
#define BATCH     2
#define SEQ       2048
#define ROWS      (BATCH * SEQ)          // 4096 token rows
#define QSZ       ((size_t)BATCH * NUM_HEADS * SEQ * HEAD_DIM)  // 4,194,304 f32

// ---------------------------------------------------------------------------
// bf16 fragment loader (fp32 source, converted in registers).
// CDNA5 16-bit A 16x32 layout: lanes 0-15 hold K=k0+[0..7] and k0+[16..23],
// lanes 16-31 hold K=k0+[8..15] and k0+[24..31]; row M = lane&15 (+r0).
// B (32x16) is the dual: per-lane column N = (lane&15)+c0, same K halves.
// Contiguous 8-float runs -> global_load_b128 pairs + v_cvt_pk_bf16_f32.
// ---------------------------------------------------------------------------
static __device__ inline v16bf frag_rows(const float* __restrict__ base, int ld,
                                         int r0, int k0) {
  const int lane = threadIdx.x & 31;
  const int r  = r0 + (lane & 15);
  const int kb = k0 + ((lane >> 4) << 3);
  const float* p = base + (size_t)r * ld + kb;
  v16bf f;
#pragma unroll
  for (int i = 0; i < 8; ++i) {
    f[i]     = (__bf16)p[i];
    f[i + 8] = (__bf16)p[i + 16];
  }
  return f;
}

static __device__ inline v8f wmma_bf16(v16bf a, v16bf b, v8f c) {
  // v_wmma_f32_16x16x32_bf16: D = A x B + C
  return __builtin_amdgcn_wmma_f32_16x16x32_bf16(false, a, false, b,
                                                 (short)0, c, false, false);
}

// ---------------------------------------------------------------------------
// Kernel 1: fused Q/K/V projection.  X[4096,1024] @ W^T[1024,1024] + bias.
// Q,K scattered into head-major [B,H,S,Hd]; V scattered TRANSPOSED into
// [B,H,Hd,S] so the attn@V GEMM gets contiguous B-fragments.
// One wave -> 32x64 output tile: per 32-deep K step, 2 A-frags are shared by
// 4 B-frags -> 8 WMMAs per 6 fragment loads.
// __launch_bounds__(128,1): lift the VGPR cap (8 f32 accs + frags + load
// temps) so nothing spills; WMMA ILP covers latency instead of occupancy.
// ---------------------------------------------------------------------------
__global__ __launch_bounds__(128, 1)
void qkv_proj_kernel(const float* __restrict__ q_in,
                     const float* __restrict__ k_in,
                     const float* __restrict__ v_in,
                     const float* __restrict__ w_q, const float* __restrict__ b_q,
                     const float* __restrict__ w_k, const float* __restrict__ b_k,
                     const float* __restrict__ w_v, const float* __restrict__ b_v,
                     float* __restrict__ Qh, float* __restrict__ Kh,
                     float* __restrict__ Vt) {
  const int which = blockIdx.z;
  const float* X  = (which == 0) ? q_in : (which == 1) ? k_in : v_in;
  const float* W  = (which == 0) ? w_q  : (which == 1) ? w_k  : w_v;
  const float* Bi = (which == 0) ? b_q  : (which == 1) ? b_k  : b_v;
  float*       O  = (which == 0) ? Qh   : (which == 1) ? Kh   : Vt;

  const int wid  = blockIdx.x * (blockDim.x >> 5) + (threadIdx.x >> 5);
  const int row0 = (wid >> 4) * 32;     // 128 row tiles of 32
  const int col0 = (wid & 15) * 64;     // 16 col tiles of 64

  v8f acc0[4] = {}, acc1[4] = {};
  for (int k0 = 0; k0 < D_MODEL; k0 += 32) {
    v16bf a0 = frag_rows(X, D_MODEL, row0,      k0);
    v16bf a1 = frag_rows(X, D_MODEL, row0 + 16, k0);
#pragma unroll
    for (int j = 0; j < 4; ++j) {
      v16bf b = frag_rows(W, D_MODEL, col0 + j * 16, k0);  // row m of W = col m of B
      acc0[j] = wmma_bf16(a0, b, acc0[j]);
      acc1[j] = wmma_bf16(a1, b, acc1[j]);
    }
  }

  const int lane = threadIdx.x & 31;
  const int nn   = lane & 15;
  const int mb   = row0 + ((lane >> 4) << 3);
#pragma unroll
  for (int t = 0; t < 2; ++t) {
    const v8f* acc = t ? acc1 : acc0;
    const int  mbt = mb + t * 16;
#pragma unroll
    for (int j = 0; j < 4; ++j) {
      const int c    = col0 + j * 16 + nn;
      const int h    = c >> 6;
      const int hd   = c & 63;
      const float bv = Bi[c];
#pragma unroll
      for (int i = 0; i < 8; ++i) {
        const int r = mbt + i;          // global token row
        const int b = r >> 11;          // / SEQ
        const int s = r & (SEQ - 1);
        const float v = acc[j][i] + bv;
        if (which == 2)                 // V: [B,H,Hd,S]
          O[(((size_t)(b * NUM_HEADS + h) * HEAD_DIM) + hd) * SEQ + s] = v;
        else                            // Q,K: [B,H,S,Hd]
          O[(((size_t)(b * NUM_HEADS + h) * SEQ) + s) * HEAD_DIM + hd] = v;
      }
    }
  }
}

// ---------------------------------------------------------------------------
// Kernel 2: scores = Q K^T / sqrt(Hd), per (b,h).  M=N=2048, K=64.
// 32x64 tile per wave.
// ---------------------------------------------------------------------------
__global__ __launch_bounds__(128, 1)
void scores_kernel(const float* __restrict__ Qh,
                   const float* __restrict__ Kh,
                   float* __restrict__ attn) {
  const int bh   = blockIdx.z;                       // 0..31
  const int wid  = blockIdx.x * (blockDim.x >> 5) + (threadIdx.x >> 5);
  const int row0 = (wid >> 5) * 32;                  // 64 row tiles of 32
  const int col0 = (wid & 31) * 64;                  // 32 col tiles of 64

  const float* Q = Qh + (size_t)bh * SEQ * HEAD_DIM;
  const float* K = Kh + (size_t)bh * SEQ * HEAD_DIM;
  float*       S = attn + (size_t)bh * SEQ * SEQ;

  v8f acc0[4] = {}, acc1[4] = {};
#pragma unroll
  for (int k0 = 0; k0 < HEAD_DIM; k0 += 32) {
    v16bf a0 = frag_rows(Q, HEAD_DIM, row0,      k0);
    v16bf a1 = frag_rows(Q, HEAD_DIM, row0 + 16, k0);
#pragma unroll
    for (int j = 0; j < 4; ++j) {
      v16bf b = frag_rows(K, HEAD_DIM, col0 + j * 16, k0);  // K row n = B col n
      acc0[j] = wmma_bf16(a0, b, acc0[j]);
      acc1[j] = wmma_bf16(a1, b, acc1[j]);
    }
  }

  const int lane = threadIdx.x & 31;
  const int nn   = lane & 15;
  const int mb   = row0 + ((lane >> 4) << 3);
  const float scale = 0.125f;                        // 1/sqrt(64)
#pragma unroll
  for (int t = 0; t < 2; ++t) {
    const v8f* acc = t ? acc1 : acc0;
    const int  mbt = mb + t * 16;
#pragma unroll
    for (int j = 0; j < 4; ++j) {
      const int c = col0 + j * 16 + nn;
#pragma unroll
      for (int i = 0; i < 8; ++i)
        S[(size_t)(mbt + i) * SEQ + c] = acc[j][i] * scale;
    }
  }
}

// ---------------------------------------------------------------------------
// Kernel 3: in-place row softmax over attn (65536 rows x 2048), fp32.
// ---------------------------------------------------------------------------
__global__ __launch_bounds__(256)
void softmax_kernel(float* __restrict__ attn) {
  float* p = attn + (size_t)blockIdx.x * SEQ;
  __shared__ float red[8];
  const int tid = threadIdx.x;

  float m = -3.4e38f;
  for (int i = tid; i < SEQ; i += 256) m = fmaxf(m, p[i]);
#pragma unroll
  for (int off = 16; off > 0; off >>= 1) m = fmaxf(m, __shfl_xor(m, off, 32));
  if ((tid & 31) == 0) red[tid >> 5] = m;
  __syncthreads();
  m = red[0];
#pragma unroll
  for (int i = 1; i < 8; ++i) m = fmaxf(m, red[i]);
  __syncthreads();

  float s = 0.0f;
  for (int i = tid; i < SEQ; i += 256) {
    float e = __expf(p[i] - m);
    p[i] = e;
    s += e;
  }
#pragma unroll
  for (int off = 16; off > 0; off >>= 1) s += __shfl_xor(s, off, 32);
  if ((tid & 31) == 0) red[tid >> 5] = s;
  __syncthreads();
  s = 0.0f;
#pragma unroll
  for (int i = 0; i < 8; ++i) s += red[i];
  const float inv = 1.0f / s;
  for (int i = tid; i < SEQ; i += 256) p[i] *= inv;
}

// ---------------------------------------------------------------------------
// Kernel 4: ctx = attn @ V per (b,h).  M=2048, K=2048, N=64.
// V is stored transposed [B,H,Hd,S], so B-frags are contiguous frag_rows.
// 32x64 tile per wave (full N).  Result scattered into [B,S,D].
// ---------------------------------------------------------------------------
__global__ __launch_bounds__(128, 1)
void av_kernel(const float* __restrict__ attn,
               const float* __restrict__ Vt,
               float* __restrict__ ctx) {
  const int bh   = blockIdx.z;
  const int wid  = blockIdx.x * (blockDim.x >> 5) + (threadIdx.x >> 5);
  const int row0 = wid * 32;                         // 64 row tiles of 32

  const float* A = attn + (size_t)bh * SEQ * SEQ;
  const float* V = Vt   + (size_t)bh * HEAD_DIM * SEQ;   // [Hd, S]
  const int b = bh >> 4, h = bh & 15;

  v8f acc0[4] = {}, acc1[4] = {};
  for (int k0 = 0; k0 < SEQ; k0 += 32) {
    v16bf a0 = frag_rows(A, SEQ, row0,      k0);
    v16bf a1 = frag_rows(A, SEQ, row0 + 16, k0);
#pragma unroll
    for (int j = 0; j < 4; ++j) {
      v16bf bf = frag_rows(V, SEQ, j * 16, k0);      // Vt row hd = B col hd
      acc0[j] = wmma_bf16(a0, bf, acc0[j]);
      acc1[j] = wmma_bf16(a1, bf, acc1[j]);
    }
  }

  const int lane = threadIdx.x & 31;
  const int nn   = lane & 15;
  const int mb   = row0 + ((lane >> 4) << 3);
#pragma unroll
  for (int t = 0; t < 2; ++t) {
    const v8f* acc = t ? acc1 : acc0;
    const int  mbt = mb + t * 16;
#pragma unroll
    for (int j = 0; j < 4; ++j) {
      const int hd = j * 16 + nn;
#pragma unroll
      for (int i = 0; i < 8; ++i) {
        const int s = mbt + i;
        ctx[((size_t)(b * SEQ + s)) * D_MODEL + h * HEAD_DIM + hd] = acc[j][i];
      }
    }
  }
}

// ---------------------------------------------------------------------------
// Kernel 5: out = ctx @ w_o^T + b_o.  [4096,1024] x [1024,1024].
// 32x64 tile per wave.
// ---------------------------------------------------------------------------
__global__ __launch_bounds__(128, 1)
void out_proj_kernel(const float* __restrict__ ctx,
                     const float* __restrict__ w_o,
                     const float* __restrict__ b_o,
                     float* __restrict__ out) {
  const int wid  = blockIdx.x * (blockDim.x >> 5) + (threadIdx.x >> 5);
  const int row0 = (wid >> 4) * 32;
  const int col0 = (wid & 15) * 64;

  v8f acc0[4] = {}, acc1[4] = {};
  for (int k0 = 0; k0 < D_MODEL; k0 += 32) {
    v16bf a0 = frag_rows(ctx, D_MODEL, row0,      k0);
    v16bf a1 = frag_rows(ctx, D_MODEL, row0 + 16, k0);
#pragma unroll
    for (int j = 0; j < 4; ++j) {
      v16bf b = frag_rows(w_o, D_MODEL, col0 + j * 16, k0);
      acc0[j] = wmma_bf16(a0, b, acc0[j]);
      acc1[j] = wmma_bf16(a1, b, acc1[j]);
    }
  }

  const int lane = threadIdx.x & 31;
  const int nn   = lane & 15;
  const int mb   = row0 + ((lane >> 4) << 3);
#pragma unroll
  for (int t = 0; t < 2; ++t) {
    const v8f* acc = t ? acc1 : acc0;
    const int  mbt = mb + t * 16;
#pragma unroll
    for (int j = 0; j < 4; ++j) {
      const int c  = col0 + j * 16 + nn;
      const float bv = b_o[c];
#pragma unroll
      for (int i = 0; i < 8; ++i)
        out[(size_t)(mbt + i) * D_MODEL + c] = acc[j][i] + bv;
    }
  }
}

// ---------------------------------------------------------------------------
extern "C" void kernel_launch(void* const* d_in, const int* in_sizes, int n_in,
                              void* d_out, int out_size, void* d_ws, size_t ws_size,
                              hipStream_t stream) {
  const float* query = (const float*)d_in[0];
  const float* key   = (const float*)d_in[1];
  const float* value = (const float*)d_in[2];
  const float* w_q   = (const float*)d_in[3];
  const float* b_q   = (const float*)d_in[4];
  const float* w_k   = (const float*)d_in[5];
  const float* b_k   = (const float*)d_in[6];
  const float* w_v   = (const float*)d_in[7];
  const float* b_v   = (const float*)d_in[8];
  const float* w_o   = (const float*)d_in[9];
  const float* b_o   = (const float*)d_in[10];

  float* out  = (float*)d_out;                        // [B,S,D]
  float* attn = out + (size_t)ROWS * D_MODEL;         // [B,H,S,S]

  float* Qh  = (float*)d_ws;                          // [B,H,S,Hd]
  float* Kh  = Qh + QSZ;                              // [B,H,S,Hd]
  float* Vt  = Kh + QSZ;                              // [B,H,Hd,S] (transposed)
  float* ctx = Vt + QSZ;                              // [B,S,D]

  // 1) QKV projections: 2048 waves per matrix (32x64 tiles), 4 waves/block
  qkv_proj_kernel<<<dim3(512, 1, 3), 128, 0, stream>>>(
      query, key, value, w_q, b_q, w_k, b_k, w_v, b_v, Qh, Kh, Vt);

  // 2) scores: 2048 waves per (b,h), z = bh
  scores_kernel<<<dim3(512, 1, 32), 128, 0, stream>>>(Qh, Kh, attn);

  // 3) softmax: one 256-thread block per attention row
  softmax_kernel<<<dim3(BATCH * NUM_HEADS * SEQ), 256, 0, stream>>>(attn);

  // 4) attn @ V: 64 waves per (b,h)
  av_kernel<<<dim3(16, 1, 32), 128, 0, stream>>>(attn, Vt, ctx);

  // 5) output projection
  out_proj_kernel<<<dim3(512), 128, 0, stream>>>(ctx, w_o, b_o, out);
}
